// ElmanLeaky_7206955123458
// MI455X (gfx1250) — matrix operand order repaired and checked by reading, not compiled, not run on hardware
//
#include <hip/hip_runtime.h>
#include <stdint.h>

// Problem constants (reference: T=1024, B=32, D=H=1024, fp32)
#define TSTEPS 1024
#define BB     32
#define DDIM   1024
#define HDIM   1024
#define KTILES 32          // K / 32
#define NTILES 64          // H / 16
#define FRAG_W 8           // u32 per lane per 16x32 f16 fragment
#define FRAG_SZ ((size_t)KTILES * NTILES * 32 * FRAG_W)   // u32 per packed half-weight (524288)
#define SCAN_WGS 32        // persistent workgroups in the scan (4 waves each)

typedef __attribute__((ext_vector_type(16))) _Float16 v16h;
typedef __attribute__((ext_vector_type(8)))  float    v8f;

union FragH { v16h h; uint4 q[2]; };

// ---------------------------------------------------------------------------
// Fragment helpers
// ---------------------------------------------------------------------------

// Load a B fragment (8 u32 = 16 f16) from a fully-resolved pointer (global or LDS).
__device__ __forceinline__ v16h load_frag_u(const uint32_t* p) {
    FragH f;
    f.q[0] = ((const uint4*)p)[0];
    f.q[1] = ((const uint4*)p)[1];
    return f.h;
}

// Load a prebuilt f16 A fragment (two contiguous 16B chunks) from an f16 array
// (global h16 state, or LDS-staged x tile).
__device__ __forceinline__ v16h load_a16(const _Float16* base) {
    FragH f;
    f.q[0] = *(const uint4*)(base);
    f.q[1] = *(const uint4*)(base + 16);
    return f.h;
}

// Split-precision triple WMMA: acc += Ahi*Bhi + Ahi*Blo + Alo*Bhi  (fp32 accumulate)
__device__ __forceinline__ v8f mma3(v8f acc, v16h ah, v16h al, v16h bh, v16h bl) {
    acc = __builtin_amdgcn_wmma_f32_16x16x32_f16(false, ah, false, bh, (short)0, acc, false, false);
    acc = __builtin_amdgcn_wmma_f32_16x16x32_f16(false, ah, false, bl, (short)0, acc, false, false);
    acc = __builtin_amdgcn_wmma_f32_16x16x32_f16(false, al, false, bh, (short)0, acc, false, false);
    return acc;
}

// Hardware tanh (gfx1250 V_TANH_F32 transcendental).
__device__ __forceinline__ float fast_tanh(float x) {
#if __has_builtin(__builtin_amdgcn_tanhf)
    return __builtin_amdgcn_tanhf(x);
#else
    float r;
    asm("v_tanh_f32 %0, %1\n\tv_nop" : "=v"(r) : "v"(x));  // v_nop: TRANS co-exec hazard
    return r;
#endif
}

// ---------------------------------------------------------------------------
// Kernel 1: pack weights [H_out=1024, K=1024] into per-lane B-fragment layout,
// split into f16 hi / f16 lo residual.  Layout (u32):
//   packed[half][(k0*64 + ntile)*256 + lane*8 + v]
// B-fragment semantics: lane L holds column N = n0 + (L&15); VGPR v holds
// K = 8*(L>>4) + 2v + (v>=4 ? 8 : 0) (lo16) and K+1 (hi16) of the 32-K chunk.
// ---------------------------------------------------------------------------
__global__ __launch_bounds__(256) void pack_weights_kernel(
        const float* __restrict__ Wx, const float* __restrict__ Wd,
        const float* __restrict__ R, uint32_t* __restrict__ dst) {
    int idx = blockIdx.x * blockDim.x + threadIdx.x;   // [0, 3<<19)
    int v    =  idx        & 7;
    int lane = (idx >> 3)  & 31;
    int nt   = (idx >> 8)  & 63;
    int k0   = (idx >> 14) & 31;
    int w    =  idx >> 19;
    if (w >= 3) return;
    const float* W = (w == 0) ? Wx : ((w == 1) ? Wd : R);
    int krow = k0 * 32 + 8 * (lane >> 4) + 2 * v + (v >= 4 ? 8 : 0);
    int n0   = nt * 16 + (lane & 15);
    float f0 = W[(size_t)n0 * HDIM + krow];
    float f1 = W[(size_t)n0 * HDIM + krow + 1];
    _Float16 h0 = (_Float16)f0, h1 = (_Float16)f1;
    _Float16 l0 = (_Float16)(f0 - (float)h0), l1 = (_Float16)(f1 - (float)h1);
    union { _Float16 h[2]; uint32_t u; } phi, plo;
    phi.h[0] = h0; phi.h[1] = h1;
    plo.h[0] = l0; plo.h[1] = l1;
    size_t fo = ((size_t)(k0 * 64 + nt) * 32 + lane) * FRAG_W + v;
    uint32_t* base = dst + (size_t)w * 2 * FRAG_SZ;
    base[fo]           = phi.u;   // hi half
    base[FRAG_SZ + fo] = plo.u;   // lo half
}

// ---------------------------------------------------------------------------
// Kernel 2: split h0 into f16 hi/lo state buffers (double-buffer slot 0).
// ---------------------------------------------------------------------------
__global__ void init_h16_kernel(const float* __restrict__ h0,
                                _Float16* __restrict__ hhi,
                                _Float16* __restrict__ hlo) {
    int i = blockIdx.x * blockDim.x + threadIdx.x;
    if (i < BB * HDIM) {
        float f = h0[i];
        _Float16 h = (_Float16)f;
        hhi[i] = h;
        hlo[i] = (_Float16)(f - (float)h);
    }
}

// ---------------------------------------------------------------------------
// Kernel 3: fused projection GEMMs.
//   wx_out    = x @ Wx^T + bias            -> written into d_out (consumed+overwritten by scan)
//   delta_out = sigmoid(x @ Wd^T + b_delta)
// 8 waves/block share one 16-row x tile: cooperatively loaded, split into
// f16 hi/lo, and staged in 64 KB LDS once; each wave then runs a 16x16
// tile-pair with split-f16 WMMA over K=1024, A-side from ds_load_b128.
// ---------------------------------------------------------------------------
__global__ __launch_bounds__(256) void proj_gemm_kernel(
        const float* __restrict__ x, const float* __restrict__ bias,
        const float* __restrict__ b_delta, const uint32_t* __restrict__ packed,
        float* __restrict__ wx_out, float* __restrict__ delta_out) {
    __shared__ _Float16 axh[16 * DDIM];   // 32 KB: x tile hi
    __shared__ _Float16 axl[16 * DDIM];   // 32 KB: x tile lo residual

    const uint32_t* WxHi = packed;
    const uint32_t* WxLo = packed + FRAG_SZ;
    const uint32_t* WdHi = packed + 2 * FRAG_SZ;
    const uint32_t* WdLo = packed + 3 * FRAG_SZ;

    int wave = threadIdx.x >> 5, lane = threadIdx.x & 31;
    int mt = blockIdx.x >> 3;                         // M-tile, shared by block
    int nt = ((blockIdx.x & 7) << 3) + wave;          // N-tile per wave
    int r0 = mt * 16;

    // Stage + split the contiguous 16x1024 x tile (16384 floats).
    const float* xt = x + (size_t)r0 * DDIM;
#pragma unroll
    for (int i = 0; i < 16; ++i) {
        int e = (i * 256 + threadIdx.x) * 4;
        float4 v4 = *(const float4*)(xt + e);
        float vv[4] = {v4.x, v4.y, v4.z, v4.w};
#pragma unroll
        for (int j = 0; j < 4; ++j) {
            _Float16 h = (_Float16)vv[j];
            axh[e + j] = h;
            axl[e + j] = (_Float16)(vv[j] - (float)h);
        }
    }
    __syncthreads();

    int m = lane & 15, g = lane >> 4;
    const _Float16* ah_base = axh + m * DDIM + 8 * g;
    const _Float16* al_base = axl + m * DDIM + 8 * g;

    v8f accx = {}, accd = {};
    for (int k0 = 0; k0 < KTILES; ++k0) {
        v16h ah = load_a16(ah_base + k0 * 32);        // ds_load_b128 x2
        v16h al = load_a16(al_base + k0 * 32);
        size_t co = (size_t)(k0 * 64 + nt) * 256 + lane * FRAG_W;
        v16h bxh = load_frag_u(WxHi + co), bxl = load_frag_u(WxLo + co);
        v16h bdh = load_frag_u(WdHi + co), bdl = load_frag_u(WdLo + co);
        accx = mma3(accx, ah, al, bxh, bxl);
        accd = mma3(accd, ah, al, bdh, bdl);
    }
    int n = nt * 16 + m;                              // C/D layout: column = lane&15
    float bs = bias[n], bd = b_delta[n];
#pragma unroll
    for (int r = 0; r < 8; ++r) {
        size_t row = (size_t)r0 + r + 8 * g;          // C/D layout: M = r + 8*(lane>>4)
        size_t o = row * HDIM + n;
        wx_out[o] = accx[r] + bs;
        float z = accd[r] + bd;
        delta_out[o] = 1.0f / (1.0f + __expf(-z));
    }
}

// ---------------------------------------------------------------------------
// Grid-wide barrier for the persistent scan: cluster barrier when launched as a
// wave-cluster (ClusterID != 0), else monotonic atomic-counter barrier.
// ---------------------------------------------------------------------------
__device__ __forceinline__ void grid_barrier(unsigned* cnt, unsigned target) {
    __syncthreads();
    if (threadIdx.x == 0) {
        __threadfence();                               // release h16 stores (device scope)
        if (__builtin_amdgcn_cluster_id_x() != 0) {
            __builtin_amdgcn_s_cluster_barrier();      // s_barrier_signal/-3 + s_barrier_wait/-3
        } else {
            atomicAdd(cnt, 1u);
            while (atomicAdd(cnt, 0u) < target) {      // relaxed L2 atomic load
                __builtin_amdgcn_s_sleep(2);
            }
        }
    }
    __syncthreads();
    __threadfence();                                   // acquire: invalidate WGP caches
}

// ---------------------------------------------------------------------------
// Kernel 4: persistent sequential scan.  32 WGs x 4 waves = 128 waves, one
// 16x16 tile of h per wave.  Each WG owns 2 N-tiles; their R fragments
// (hi+lo) live in 128 KB LDS (CU-mode safe).  h state is double-buffered
// f16 hi/lo in global (L2-resident); each wave keeps its own fp32 h slice in
// registers for the leaky blend.
//   cand  = tanh(h @ R^T + wx_t)       (hardware v_tanh_f32)
//   h_new = (1-delta)*h + delta*cand   (written over wx_t in d_out)
// ---------------------------------------------------------------------------
__global__ __launch_bounds__(128) void scan_kernel(
        const float* __restrict__ h0, const uint32_t* __restrict__ Rpacked,
        const float* __restrict__ delta, float* __restrict__ out,
        _Float16* __restrict__ h16, unsigned* __restrict__ counter) {
    extern __shared__ uint32_t lds[];                 // 128 chunks x 256 u32 = 128 KB
    const uint32_t* RHi = Rpacked;
    const uint32_t* RLo = Rpacked + FRAG_SZ;

    int g    = blockIdx.x;                            // 32 WGs
    int wave = threadIdx.x >> 5, lane = threadIdx.x & 31;
    int ntl  = wave >> 1;                             // local N-tile 0..1
    int nt   = 2 * g + ntl;                           // global N-tile 0..63
    int mt   = wave & 1;                              // M-tile 0..1 (B=32)

    // Preload this WG's R fragments into LDS: chunk c = ((half*32+k0)*2 + ntl)
    for (int c = 0; c < 128; ++c) {
        int half = c >> 6, k0 = (c >> 1) & 31, l = c & 1;
        const uint32_t* src = (half ? RLo : RHi) + (size_t)(k0 * 64 + 2 * g + l) * 256;
#pragma unroll
        for (int j = 0; j < 2; ++j)
            lds[(size_t)c * 256 + j * 128 + threadIdx.x] = src[j * 128 + threadIdx.x];
    }
    __syncthreads();

    int mlow = lane & 15, gg = lane >> 4;
    int n    = nt * 16 + mlow;                        // C/D column
    int m0   = mt * 16;
    int arow = m0 + mlow;                             // A-fragment row

    v8f hprev;
#pragma unroll
    for (int r = 0; r < 8; ++r)
        hprev[r] = h0[(size_t)(m0 + r + 8 * gg) * HDIM + n];

    const size_t HS = (size_t)BB * HDIM;              // 32768 elements
    unsigned nblocks = gridDim.x;

    for (int t = 0; t < TSTEPS; ++t) {
        int cur = t & 1;
        const _Float16* hhi = h16 + (size_t)cur * 2 * HS;
        const _Float16* hlo = hhi + HS;
        _Float16* nhi = h16 + (size_t)(cur ^ 1) * 2 * HS;
        _Float16* nlo = nhi + HS;

        const _Float16* ah_base = hhi + (size_t)arow * HDIM + 8 * gg;
        const _Float16* al_base = hlo + (size_t)arow * HDIM + 8 * gg;

        v8f acc = {};
        for (int k0 = 0; k0 < KTILES; ++k0) {
            v16h ah = load_a16(ah_base + k0 * 32);
            v16h al = load_a16(al_base + k0 * 32);
            size_t lo_ = ((size_t)(k0 * 2 + ntl) * 32 + lane) * FRAG_W;
            v16h bh = load_frag_u(lds + lo_);                     // hi half (chunks 0..63)
            v16h bl = load_frag_u(lds + (size_t)64 * 256 + lo_);  // lo half (chunks 64..127)
            acc = mma3(acc, ah, al, bh, bl);
        }

        size_t ob = (size_t)t * HS;
#pragma unroll
        for (int r = 0; r < 8; ++r) {
            int row = m0 + r + 8 * gg;
            size_t o = ob + (size_t)row * HDIM + n;
            float wx = out[o];                        // wx_t (written by proj_gemm)
            float dl = delta[o];
            float cand = fast_tanh(acc[r] + wx);
            float hn = (1.0f - dl) * hprev[r] + dl * cand;
            hprev[r] = hn;
            out[o] = hn;                              // overwrite wx_t with h_t
            _Float16 hh = (_Float16)hn;
            size_t ho = (size_t)row * HDIM + n;
            nhi[ho] = hh;
            nlo[ho] = (_Float16)(hn - (float)hh);
        }

        if (t + 1 < TSTEPS) {                         // warm L2 lines for next step
            size_t p = ob + HS + (size_t)(m0 + 8 * gg) * HDIM + n;
            __builtin_prefetch(out + p, 0, 1);
            __builtin_prefetch(delta + p, 0, 1);
        }

        grid_barrier(counter, (unsigned)(t + 1) * nblocks);
    }
}

// ---------------------------------------------------------------------------
// Launch: zero barrier counter -> pack weights -> init h16 -> projections ->
// persistent scan.  All stream-ordered; graph-capture safe (async memset only).
// Workspace: delta 128MB | packed weights 12MB | h16 256KB | counter.
// ---------------------------------------------------------------------------
extern "C" void kernel_launch(void* const* d_in, const int* in_sizes, int n_in,
                              void* d_out, int out_size, void* d_ws, size_t ws_size,
                              hipStream_t stream) {
    const float* x    = (const float*)d_in[0];
    const float* h0   = (const float*)d_in[1];
    const float* Wx   = (const float*)d_in[2];
    const float* R    = (const float*)d_in[3];
    const float* bias = (const float*)d_in[4];
    const float* Wd   = (const float*)d_in[5];
    const float* bd   = (const float*)d_in[6];
    float* out = (float*)d_out;

    char* ws = (char*)d_ws;
    const size_t TBH = (size_t)TSTEPS * BB * HDIM;

    float* delta = (float*)ws;
    size_t off = TBH * sizeof(float);
    uint32_t* packed = (uint32_t*)(ws + off);
    off += 6ull * FRAG_SZ * sizeof(uint32_t);
    _Float16* h16 = (_Float16*)(ws + off);            // 2 buffers x (hi,lo) x B*H
    off += 4ull * BB * HDIM * sizeof(_Float16);
    off = (off + 255) & ~(size_t)255;
    unsigned* counter = (unsigned*)(ws + off);

    hipMemsetAsync(counter, 0, sizeof(unsigned), stream);

    pack_weights_kernel<<<(3u << 19) / 256, 256, 0, stream>>>(Wx, Wd, R, packed);
    init_h16_kernel<<<(BB * HDIM) / 256, 256, 0, stream>>>(
        h0, h16, h16 + (size_t)BB * HDIM);
    proj_gemm_kernel<<<16384, 256, 0, stream>>>(x, bias, bd, packed, out, delta);

    const uint32_t* Rpacked = packed + 4ull * FRAG_SZ;
    scan_kernel<<<SCAN_WGS, 128, 128 * 1024, stream>>>(h0, Rpacked, delta, out, h16, counter);
}
